// Decoder_46531675685089
// MI455X (gfx1250) — matrix operand order repaired and checked by reading, not compiled, not run on hardware
//
#include <hip/hip_runtime.h>
#include <hip/hip_bf16.h>

// ---------------------------------------------------------------------------
// CDNA5 (gfx1250) WMMA transformer decoder.
// All GEMMs: v_wmma_f32_16x16x32_f16, fp32 -> f16 staged through LDS with
// double buffering, b128 global loads, v_cvt_pk_rtz_f16_f32 packing.
// wave32, 256 threads/block = 8 waves, tile 128x64x32.
// ---------------------------------------------------------------------------

typedef __attribute__((ext_vector_type(16))) _Float16 v16h;
typedef __attribute__((ext_vector_type(2)))  _Float16 v2h;
typedef __attribute__((ext_vector_type(8)))  float    v8f;

#define TM 128
#define TN 64
#define TK 32
#define LROW 34                 // padded LDS row stride in halves (68B = 17 banks)

// pack two f32 -> packed f16 pair (v_cvt_pk_rtz_f16_f32)
__device__ __forceinline__ v2h pk2(float x, float y)
{
    return __builtin_bit_cast(v2h, __builtin_amdgcn_cvt_pkrtz(x, y));
}

// ---- staging helpers (guard-free: all launch dims are exact tile multiples) ----

template <bool AHEADS>
__device__ __forceinline__ void stageA(_Float16* __restrict__ As,
                                       const float* __restrict__ A, int lda,
                                       int mBase, int k0, int tid)
{
    // 128x32 tile = 1024 float4, 4 per thread (8 float4 per row)
#pragma unroll
    for (int it = 0; it < 4; ++it) {
        const int j  = tid + it * 256;
        const int r  = j >> 3;
        const int c4 = j & 7;
        size_t idx;
        if (AHEADS)  // A[(k>>6)*lda + m*64 + (k&63)], k-tile never crosses a 64-block
            idx = (size_t)(k0 >> 6) * (size_t)lda + (size_t)(mBase + r) * 64
                + (size_t)((k0 & 63) + c4 * 4);
        else
            idx = (size_t)(mBase + r) * (size_t)lda + (size_t)(k0 + c4 * 4);
        const float4 v = *(const float4*)(A + idx);
        v2h* dst = (v2h*)(As + r * LROW + c4 * 4);
        dst[0] = pk2(v.x, v.y);
        dst[1] = pk2(v.z, v.w);
    }
}

template <bool BT>
__device__ __forceinline__ void stageB(_Float16* __restrict__ Bs,
                                       const float* __restrict__ Bm, int ldb,
                                       int nBase, int k0, int tid)
{
    if (BT) {
        // B is (N,K) row-major: contiguous along k. 64x32 = 512 float4, 2/thread.
#pragma unroll
        for (int it = 0; it < 2; ++it) {
            const int j  = tid + it * 256;
            const int n  = j >> 3;
            const int c4 = j & 7;
            const float4 v = *(const float4*)(Bm + (size_t)(nBase + n) * ldb
                                                 + (size_t)(k0 + c4 * 4));
            v2h* dst = (v2h*)(Bs + n * LROW + c4 * 4);
            dst[0] = pk2(v.x, v.y);
            dst[1] = pk2(v.z, v.w);
        }
    } else {
        // B is (K,N) row-major: contiguous along n; scatter-transpose into [n][k].
#pragma unroll
        for (int it = 0; it < 2; ++it) {
            const int j  = tid + it * 256;
            const int k  = j >> 4;
            const int n4 = j & 15;
            const float4 v = *(const float4*)(Bm + (size_t)(k0 + k) * ldb
                                                 + (size_t)(nBase + n4 * 4));
            Bs[(n4 * 4 + 0) * LROW + k] = (_Float16)v.x;
            Bs[(n4 * 4 + 1) * LROW + k] = (_Float16)v.y;
            Bs[(n4 * 4 + 2) * LROW + k] = (_Float16)v.z;
            Bs[(n4 * 4 + 3) * LROW + k] = (_Float16)v.w;
        }
    }
}

// ---- fragment loads in contiguous k order (k = 16*(lane/16) + s) for A (16x32) / B (32x16) ----

__device__ __forceinline__ v16h loadAfrag(const _Float16* __restrict__ As,
                                          int am, int lane)
{
    const v2h* ap = (const v2h*)(As + am * LROW);
    const int  b  = (lane & 16) ? 8 : 0;   // K offset 16 halves for lanes 16-31
    v16h a;
#pragma unroll
    for (int p = 0; p < 4; ++p) {
        v2h u = ap[b + p];      a[2 * p]     = u[0]; a[2 * p + 1] = u[1];
        v2h w = ap[4 + b + p];  a[8 + 2 * p] = w[0]; a[9 + 2 * p] = w[1];
    }
    return a;
}

__device__ __forceinline__ v16h loadBfrag(const _Float16* __restrict__ Bs,
                                          int bn, int lane)
{
    const v2h* bp = (const v2h*)(Bs + bn * LROW) + ((lane & 16) ? 8 : 0);
    v16h b;
#pragma unroll
    for (int p = 0; p < 8; ++p) {
        v2h u = bp[p]; b[2 * p] = u[0]; b[2 * p + 1] = u[1];
    }
    return b;
}

// Generic GEMM:  C[z] = act( scale * A[z] x B[z] + bias[z] + resid )
template <bool BT, bool RELU, bool AHEADS>
__global__ void __launch_bounds__(256)
wmma_gemm(const float* __restrict__ A, size_t aStride, int lda,
          const float* __restrict__ Bm, size_t bStride, int ldb,
          const float* __restrict__ bias, size_t biasStride,
          const float* __restrict__ resid, int ldr,
          float* __restrict__ C, size_t cStride, int ldc,
          int M, int N, int K, float scale)
{
    __shared__ _Float16 As[2][TM * LROW];
    __shared__ _Float16 Bs[2][TN * LROW];

    const int tid  = threadIdx.x;
    const int lane = tid & 31;
    const int wave = tid >> 5;
    const int z    = blockIdx.z;

    A  += (size_t)z * aStride;
    Bm += (size_t)z * bStride;
    if (bias) bias += (size_t)z * biasStride;
    C  += (size_t)z * cStride;

    const int mBase = blockIdx.y * TM;
    const int nBase = blockIdx.x * TN;

    v8f acc[4] = {};

    stageA<AHEADS>(As[0], A, lda, mBase, 0, tid);
    stageB<BT>(Bs[0], Bm, ldb, nBase, 0, tid);
    __syncthreads();

    int buf = 0;
    for (int k0 = 0; k0 < K; k0 += TK) {
        const int kn = k0 + TK;
        if (kn < K) {   // stage next tile into the other buffer (overlaps WMMA)
            stageA<AHEADS>(As[buf ^ 1], A, lda, mBase, kn, tid);
            stageB<BT>(Bs[buf ^ 1], Bm, ldb, nBase, kn, tid);
        }

        const int am = wave * 16 + (lane & 15);
        const v16h af = loadAfrag(As[buf], am, lane);
#pragma unroll
        for (int j = 0; j < 4; ++j) {
            const v16h bf = loadBfrag(Bs[buf], j * 16 + (lane & 15), lane);
            acc[j] = __builtin_amdgcn_wmma_f32_16x16x32_f16(
                         false, af, false, bf, (short)0, acc[j], false, false);
        }
        __syncthreads();
        buf ^= 1;
    }

    // ---- epilogue: scale, bias, residual, relu ----
    const int mrow0 = mBase + wave * 16 + ((lane >> 4) << 3);
#pragma unroll
    for (int j = 0; j < 4; ++j) {
        const int n = nBase + j * 16 + (lane & 15);
        const float bv = bias ? bias[n] : 0.f;
#pragma unroll
        for (int r = 0; r < 8; ++r) {
            const int m = mrow0 + r;
            float v = acc[j][r] * scale + bv;
            if (resid) v += resid[(size_t)m * ldr + n];
            if (RELU) v = v > 0.f ? v : 0.f;
            C[(size_t)m * ldc + n] = v;
        }
    }
}

// Row softmax with static prefix mask (cols >= mlen -> 0), float4 I/O.
__global__ void __launch_bounds__(256)
softmax_rows(float* __restrict__ s, const int* __restrict__ maskp, int useMask, int T)
{
    __shared__ float red[256];
    const int tid = threadIdx.x;
    float4* p4 = (float4*)(s + (size_t)blockIdx.x * T);
    const int T4 = T >> 2;
    const int mlen = useMask ? *maskp : T;

    float mx = -3.4e38f;
    for (int i = tid; i < T4; i += 256) {
        const float4 v = p4[i];
        const int t = i * 4;
        if (t + 0 < mlen) mx = fmaxf(mx, v.x);
        if (t + 1 < mlen) mx = fmaxf(mx, v.y);
        if (t + 2 < mlen) mx = fmaxf(mx, v.z);
        if (t + 3 < mlen) mx = fmaxf(mx, v.w);
    }
    red[tid] = mx; __syncthreads();
    for (int st = 128; st > 0; st >>= 1) {
        if (tid < st) red[tid] = fmaxf(red[tid], red[tid + st]);
        __syncthreads();
    }
    mx = red[0]; __syncthreads();

    float sum = 0.f;
    for (int i = tid; i < T4; i += 256) {
        float4 v = p4[i];
        const int t = i * 4;
        v.x = (t + 0 < mlen) ? __expf(v.x - mx) : 0.f;
        v.y = (t + 1 < mlen) ? __expf(v.y - mx) : 0.f;
        v.z = (t + 2 < mlen) ? __expf(v.z - mx) : 0.f;
        v.w = (t + 3 < mlen) ? __expf(v.w - mx) : 0.f;
        sum += v.x + v.y + v.z + v.w;
        p4[i] = v;
    }
    red[tid] = sum; __syncthreads();
    for (int st = 128; st > 0; st >>= 1) {
        if (tid < st) red[tid] += red[tid + st];
        __syncthreads();
    }
    const float inv = 1.f / (red[0] + 1e-30f);
    for (int i = tid; i < T4; i += 256) {
        float4 v = p4[i];
        v.x *= inv; v.y *= inv; v.z *= inv; v.w *= inv;
        p4[i] = v;
    }
}

// Reference BatchNorm1d: channel axis is S; stats over (B, D). float4 I/O.
__global__ void __launch_bounds__(256)
bn_rows(const float* __restrict__ x, const float* __restrict__ g,
        const float* __restrict__ be, float* __restrict__ y,
        int Bn, int S, int D)
{
    __shared__ float rs[256], rq[256];
    const int s = blockIdx.x, tid = threadIdx.x;
    const int D4 = D >> 2;
    const int n = Bn * D;

    float sum = 0.f, sq = 0.f;
    for (int b = 0; b < Bn; ++b) {
        const float4* x4 = (const float4*)(x + ((size_t)b * S + s) * D);
        for (int d = tid; d < D4; d += 256) {
            const float4 v = x4[d];
            sum += v.x + v.y + v.z + v.w;
            sq  += v.x * v.x + v.y * v.y + v.z * v.z + v.w * v.w;
        }
    }
    rs[tid] = sum; rq[tid] = sq; __syncthreads();
    for (int st = 128; st > 0; st >>= 1) {
        if (tid < st) { rs[tid] += rs[tid + st]; rq[tid] += rq[tid + st]; }
        __syncthreads();
    }
    const float mean = rs[0] / n;
    const float var  = rq[0] / n - mean * mean;
    const float inv  = rsqrtf(var + 1e-5f);
    const float gs = g[s], bs = be[s];
    for (int b = 0; b < Bn; ++b) {
        const float4* x4 = (const float4*)(x + ((size_t)b * S + s) * D);
        float4* y4 = (float4*)(y + ((size_t)b * S + s) * D);
        for (int d = tid; d < D4; d += 256) {
            float4 v = x4[d];
            v.x = gs * (v.x - mean) * inv + bs;
            v.y = gs * (v.y - mean) * inv + bs;
            v.z = gs * (v.z - mean) * inv + bs;
            v.w = gs * (v.w - mean) * inv + bs;
            y4[d] = v;
        }
    }
}

// ---------------------------------------------------------------------------
// Host-side orchestration
// ---------------------------------------------------------------------------
namespace {
constexpr int Lc = 4, Bn = 2, Sd = 1024, Dd = 1024, Hh = 16, DKd = 64, Ff = 4096;
constexpr int BSd = Bn * Sd;

void mha(hipStream_t stream, const float* inp,
         const float* Wq, const float* bq, const float* Wk, const float* bk,
         const float* Wv, const float* bv, const float* Wo, const float* bo,
         const float* resid, float* outBuf,
         float* qb, float* kb, float* vb, float* sb, float* ab,
         const int* maskp, int useMask)
{
    const dim3 blk(256);
    // Q/K/V projections: per head z, (BS x D) x (D x 64)
    const dim3 gq(1, BSd / TM, Hh);
    wmma_gemm<false, false, false><<<gq, blk, 0, stream>>>(
        inp, 0, Dd, Wq, (size_t)Dd * DKd, DKd, bq, (size_t)DKd,
        nullptr, 0, qb, (size_t)BSd * DKd, DKd, BSd, DKd, Dd, 1.f);
    wmma_gemm<false, false, false><<<gq, blk, 0, stream>>>(
        inp, 0, Dd, Wk, (size_t)Dd * DKd, DKd, bk, (size_t)DKd,
        nullptr, 0, kb, (size_t)BSd * DKd, DKd, BSd, DKd, Dd, 1.f);
    wmma_gemm<false, false, false><<<gq, blk, 0, stream>>>(
        inp, 0, Dd, Wv, (size_t)Dd * DKd, DKd, bv, (size_t)DKd,
        nullptr, 0, vb, (size_t)BSd * DKd, DKd, BSd, DKd, Dd, 1.f);

    // scores = Q * K^T / sqrt(64); batch z = h*Bn + b (uniform strides)
    wmma_gemm<true, false, false><<<dim3(Sd / TN, Sd / TM, Hh * Bn), blk, 0, stream>>>(
        qb, (size_t)Sd * DKd, DKd, kb, (size_t)Sd * DKd, DKd,
        nullptr, 0, nullptr, 0,
        sb, (size_t)Sd * Sd, Sd, Sd, Sd, DKd, 0.125f);

    softmax_rows<<<Hh * Bn * Sd, blk, 0, stream>>>(sb, maskp, useMask, Sd);

    // attn = P * V  -> per-head layout (H, B*S, 64)
    wmma_gemm<false, false, false><<<dim3(1, Sd / TM, Hh * Bn), blk, 0, stream>>>(
        sb, (size_t)Sd * Sd, Sd, vb, (size_t)Sd * DKd, DKd,
        nullptr, 0, nullptr, 0,
        ab, (size_t)Sd * DKd, DKd, Sd, DKd, Sd, 1.f);

    // output projection over concatenated heads (AHEADS gather) + residual
    wmma_gemm<false, false, true><<<dim3(Dd / TN, BSd / TM, 1), blk, 0, stream>>>(
        ab, 0, BSd * DKd /* head stride */, Wo, 0, Dd, bo, 0,
        resid, Dd, outBuf, 0, Dd, BSd, Dd, Hh * DKd, 1.f);
}
} // namespace

extern "C" void kernel_launch(void* const* d_in, const int* in_sizes, int n_in,
                              void* d_out, int out_size, void* d_ws, size_t ws_size,
                              hipStream_t stream)
{
    (void)in_sizes; (void)n_in; (void)out_size; (void)ws_size;

    const float* x     = (const float*)d_in[0];
    const float* Wq[2] = {(const float*)d_in[1],  (const float*)d_in[9]};
    const float* bq[2] = {(const float*)d_in[2],  (const float*)d_in[10]};
    const float* Wk[2] = {(const float*)d_in[3],  (const float*)d_in[11]};
    const float* bk[2] = {(const float*)d_in[4],  (const float*)d_in[12]};
    const float* Wv[2] = {(const float*)d_in[5],  (const float*)d_in[13]};
    const float* bv[2] = {(const float*)d_in[6],  (const float*)d_in[14]};
    const float* Wo[2] = {(const float*)d_in[7],  (const float*)d_in[15]};
    const float* bo[2] = {(const float*)d_in[8],  (const float*)d_in[16]};
    const float* g1  = (const float*)d_in[17]; const float* be1 = (const float*)d_in[18];
    const float* g2  = (const float*)d_in[19]; const float* be2 = (const float*)d_in[20];
    const float* g3  = (const float*)d_in[21]; const float* be3 = (const float*)d_in[22];
    const float* W1  = (const float*)d_in[23]; const float* bf1 = (const float*)d_in[24];
    const float* W2  = (const float*)d_in[25]; const float* bf2 = (const float*)d_in[26];
    const int*   maskp = (const int*)d_in[27];

    float* w = (float*)d_ws;
    size_t off = 0;
    auto alloc = [&](size_t elems) { float* p = w + off; off += elems; return p; };
    float* qb   = alloc((size_t)Hh * BSd * DKd);
    float* kb   = alloc((size_t)Hh * BSd * DKd);
    float* vb   = alloc((size_t)Hh * BSd * DKd);
    float* sb   = alloc((size_t)Hh * Bn * Sd * Sd);
    float* ab   = alloc((size_t)Hh * BSd * DKd);
    float* x1   = alloc((size_t)BSd * Dd);
    float* yb   = alloc((size_t)BSd * Dd);
    float* hid  = alloc((size_t)BSd * Ff);
    float* tmp  = alloc((size_t)BSd * Dd);
    float* ping = alloc((size_t)BSd * Dd);
    float* pong = alloc((size_t)BSd * Dd);
    float* outf = (float*)d_out;

    const dim3 blk(256);
    const float* cur = x;
    for (int l = 0; l < Lc; ++l) {
        const size_t oW  = (size_t)l * Hh * Dd * DKd;
        const size_t oBq = (size_t)l * Hh * DKd;
        const size_t oWo = (size_t)l * (Hh * DKd) * Dd;
        const size_t oD  = (size_t)l * Dd;
        const size_t oF  = (size_t)l * Ff;

        // x1 = BN1( h + MHA1(h, masked) )
        mha(stream, cur, Wq[0] + oW, bq[0] + oBq, Wk[0] + oW, bk[0] + oBq,
            Wv[0] + oW, bv[0] + oBq, Wo[0] + oWo, bo[0] + oD,
            cur, tmp, qb, kb, vb, sb, ab, maskp, 1);
        bn_rows<<<Sd, blk, 0, stream>>>(tmp, g1 + oD, be1 + oD, x1, Bn, Sd, Dd);

        // y = BN2( h + MHA2(x1, unmasked) )   (residual from block input h)
        mha(stream, x1, Wq[1] + oW, bq[1] + oBq, Wk[1] + oW, bk[1] + oBq,
            Wv[1] + oW, bv[1] + oBq, Wo[1] + oWo, bo[1] + oD,
            cur, tmp, qb, kb, vb, sb, ab, maskp, 0);
        bn_rows<<<Sd, blk, 0, stream>>>(tmp, g2 + oD, be2 + oD, yb, Bn, Sd, Dd);

        // FFN: hid = relu(y @ W1 + bf1); out = x1 + hid @ W2 + bf2; BN3
        wmma_gemm<false, true, false><<<dim3(Ff / TN, BSd / TM, 1), blk, 0, stream>>>(
            yb, 0, Dd, W1 + (size_t)l * Dd * Ff, 0, Ff, bf1 + oF, 0,
            nullptr, 0, hid, 0, Ff, BSd, Ff, Dd, 1.f);
        wmma_gemm<false, false, false><<<dim3(Dd / TN, BSd / TM, 1), blk, 0, stream>>>(
            hid, 0, Ff, W2 + (size_t)l * Ff * Dd, 0, Dd, bf2 + oD, 0,
            x1, Dd, tmp, 0, Dd, BSd, Dd, Ff, 1.f);

        float* next = (l == Lc - 1) ? outf : ((l & 1) ? pong : ping);
        bn_rows<<<Sd, blk, 0, stream>>>(tmp, g3 + oD, be3 + oD, next, Bn, Sd, Dd);
        cur = next;
    }
}
